// LSTM_24472723652770
// MI455X (gfx1250) — compile-verified, matrix operand order checked by reading, not run on hardware
//
#include <hip/hip_runtime.h>

typedef _Float16 v16h __attribute__((ext_vector_type(16)));
typedef _Float16 v8h  __attribute__((ext_vector_type(8)));
typedef float    v8f  __attribute__((ext_vector_type(8)));

union Frag16 { v16h v; v8h h[2]; };

// Lane fragment for v_wmma_f32_16x16x32_f16 (16-bit A/B 16x32 layout,
// ISA 7.12.2): lanes 0-15 hold K=[0..7],[16..23], lanes 16-31 hold
// K=[8..15],[24..31] of their row -> two contiguous 16B loads per lane.
__device__ __forceinline__ v16h load_frag(const _Float16* rowk, int hi) {
    Frag16 f;
    f.h[0] = *(const v8h*)(rowk + hi * 8);
    f.h[1] = *(const v8h*)(rowk + 16 + hi * 8);
    return f.v;
}

__global__ void cvt_f32_to_f16(const float* __restrict__ src,
                               _Float16* __restrict__ dst, int n) {
    int i = blockIdx.x * blockDim.x + threadIdx.x;
    int stride = gridDim.x * blockDim.x;
    for (; i < n; i += stride) dst[i] = (_Float16)src[i];
}

__global__ void zero_u32(unsigned* p) { *p = 0u; }

// ---------------------------------------------------------------------------
// Bulk GEMM: C[M,N](f16) = A[M,K] * B[N,K]^T + bias[N].
// 8 waves/block; wave w owns rows [bx*128 + w*16, +16), cols [by*64, +64)
// as 4 accumulators with the A-fragment reused x4. M%128==N%64==K%32==0.
// ---------------------------------------------------------------------------
__global__ __launch_bounds__(256)
void gemm_f16_wmma(const _Float16* __restrict__ A,
                   const _Float16* __restrict__ B,
                   const float* __restrict__ bias,
                   _Float16* __restrict__ C,
                   int M, int N, int K) {
    const int lane = threadIdx.x & 31;
    const int wave = threadIdx.x >> 5;
    const int hi   = lane >> 4;
    const int lm   = lane & 15;
    const int row0 = blockIdx.x * 128 + wave * 16;
    const int col0 = blockIdx.y * 64;

    v8f acc[4] = {v8f{}, v8f{}, v8f{}, v8f{}};

    const _Float16* arow = A + (size_t)(row0 + lm) * K;
    const _Float16* brow[4];
#pragma unroll
    for (int c = 0; c < 4; ++c)
        brow[c] = B + (size_t)(col0 + c * 16 + lm) * K;

    for (int kk = 0; kk < K; kk += 32) {
        __builtin_prefetch(arow + kk + 128, 0, 1);
        v16h a = load_frag(arow + kk, hi);
#pragma unroll
        for (int c = 0; c < 4; ++c) {
            v16h b = load_frag(brow[c] + kk, hi);
            acc[c] = __builtin_amdgcn_wmma_f32_16x16x32_f16(
                false, a, false, b, (short)0, acc[c], false, false);
        }
    }

    // C/D layout: VGPR r -> lanes 0-15 M=r, lanes 16-31 M=8+r; N = lane%16.
#pragma unroll
    for (int c = 0; c < 4; ++c) {
        const int n = col0 + c * 16 + lm;
        const float bv = bias[n];
#pragma unroll
        for (int r = 0; r < 8; ++r) {
            const int m = row0 + hi * 8 + r;
            C[(size_t)m * N + n] = (_Float16)(acc[c][r] + bv);
        }
    }
}

// ---------------------------------------------------------------------------
// Persistent LSTM scan: one launch runs all T timesteps of a layer.
// Grid: H/16 = 64 blocks (block owns cols [j0, j0+16) of H); 1024 thr = 32
// waves. Wave w = (ks = w&1, gate g = (w>>1)&3, row-tile rt = w>>3): one
// 16x16 tile over half of K, two interleaved accumulators (dep chain = 8).
// W_hh slice (64x1024 f16, rows padded to 1032) staged once in LDS; c state
// is LDS-resident across all steps. Steps separated by a device-wide
// arrive/spin barrier on a monotonic atomic counter (64 blocks, 1 WG/WGP).
// ---------------------------------------------------------------------------
#define LSTM_T 512
#define LSTM_B 64
#define LSTM_H 1024
#define WPAD   1032              // 1024 + 8 halfs: kills 2KB-stride LDS bank conflicts
#define SMEM_BYTES (64 * WPAD * 2 + 2 * 4 * 64 * 16 * 4 + 64 * 16 * 4)

__global__ __launch_bounds__(1024)
void lstm_scan_persistent(const _Float16* __restrict__ gx,      // [T,B,4H] f16
                          const _Float16* __restrict__ Whh,     // [4H,H]  f16
                          const float* __restrict__ c0_l,       // [B,H]   f32
                          _Float16* __restrict__ hb0,           // [B,H] ping (holds h0 f16)
                          _Float16* __restrict__ hb1,           // [B,H] pong
                          float* __restrict__ c_n,              // [B,H] f32 (final c)
                          float* __restrict__ h_n,              // [B,H] f32 (final h)
                          float* __restrict__ out_f32,          // [T,B,H] or null
                          _Float16* __restrict__ seq_f16,       // [T,B,H] or null
                          unsigned* __restrict__ bar, int T) {
    extern __shared__ __align__(16) char smem[];
    _Float16* Wlds = (_Float16*)smem;                            // [64][WPAD]
    float*    exch = (float*)(smem + 64 * WPAD * 2);             // [2][4][64][16]
    float*    cbuf = exch + 2 * 4 * 64 * 16;                     // [64][16]

    const int tid  = threadIdx.x;
    const int lane = tid & 31;
    const int wave = tid >> 5;
    const int hi   = lane >> 4;
    const int lm   = lane & 15;
    const int ks   = wave & 1;
    const int g    = (wave >> 1) & 3;
    const int rt   = wave >> 3;
    const int j0   = blockIdx.x * 16;

    // Stage this block's W_hh slice: local row lr = g*16+rr  <-  Whh[g*H+j0+rr].
    for (int v = tid; v < 64 * 128; v += 1024) {                 // 128 v8h per row
        const int lr = v >> 7, k8 = (v & 127) * 8;
        const int grow = (lr >> 4) * LSTM_H + j0 + (lr & 15);
        *(v8h*)(Wlds + (size_t)lr * WPAD + k8) =
            *(const v8h*)(Whh + (size_t)grow * LSTM_H + k8);
    }
    // Resident cell state.
    for (int v = tid; v < LSTM_B * 16; v += 1024)
        cbuf[v] = c0_l[(size_t)(v >> 4) * LSTM_H + j0 + (v & 15)];
    __syncthreads();

    const int erow = tid >> 4, ecol = tid & 15;                  // elementwise cell
    const int ecg  = j0 + ecol;

    for (int t = 0; t < T; ++t) {
        const _Float16* hin  = (t & 1) ? hb1 : hb0;
        _Float16*       hout = (t & 1) ? hb0 : hb1;

        // --- GEMM tile: h(16x512 half-K) x Wlds^T ---
        v8f acc0 = v8f{}, acc1 = v8f{};
        const _Float16* arow = hin + (size_t)(rt * 16 + lm) * LSTM_H + ks * 512;
        const _Float16* wrow = Wlds + (size_t)(g * 16 + lm) * WPAD + ks * 512;
#pragma unroll
        for (int kk = 0; kk < 512; kk += 64) {
            v16h a0 = load_frag(arow + kk, hi);
            v16h b0 = load_frag(wrow + kk, hi);
            acc0 = __builtin_amdgcn_wmma_f32_16x16x32_f16(
                false, a0, false, b0, (short)0, acc0, false, false);
            v16h a1 = load_frag(arow + kk + 32, hi);
            v16h b1 = load_frag(wrow + kk + 32, hi);
            acc1 = __builtin_amdgcn_wmma_f32_16x16x32_f16(
                false, a1, false, b1, (short)0, acc1, false, false);
        }
#pragma unroll
        for (int r = 0; r < 8; ++r)
            exch[(((ks * 4 + g) * 64) + rt * 16 + hi * 8 + r) * 16 + lm] =
                acc0[r] + acc1[r];
        __syncthreads();

        // --- Fused gates: one cell per thread ---
        {
            const _Float16* gx_t = gx + (size_t)t * LSTM_B * 4 * LSTM_H;
            const size_t gxr = (size_t)erow * (4 * LSTM_H) + ecg;
#define EX(K, G) exch[((((K) * 4 + (G)) * 64) + erow) * 16 + ecol]
            const float pi = EX(0, 0) + EX(1, 0) + (float)gx_t[gxr + 0 * LSTM_H];
            const float pf = EX(0, 1) + EX(1, 1) + (float)gx_t[gxr + 1 * LSTM_H];
            const float pg = EX(0, 2) + EX(1, 2) + (float)gx_t[gxr + 2 * LSTM_H];
            const float po = EX(0, 3) + EX(1, 3) + (float)gx_t[gxr + 3 * LSTM_H];
#undef EX
            const float i_ = 1.0f / (1.0f + __expf(-pi));
            const float f_ = 1.0f / (1.0f + __expf(-pf));
            const float g_ = tanhf(pg);
            const float o_ = 1.0f / (1.0f + __expf(-po));
            const float c_new = f_ * cbuf[tid] + i_ * g_;
            const float h_new = o_ * tanhf(c_new);
            cbuf[tid] = c_new;
            const size_t off = (size_t)erow * LSTM_H + ecg;
            hout[off] = (_Float16)h_new;
            if (out_f32) out_f32[(size_t)t * LSTM_B * LSTM_H + off] = h_new;
            if (seq_f16) seq_f16[(size_t)t * LSTM_B * LSTM_H + off] = (_Float16)h_new;
            if (t == T - 1) {
                h_n[off] = h_new;
                c_n[off] = c_new;
            }
        }

        // --- Device-wide step barrier (monotonic counter, 64 arrivals/step) ---
        if (tid == 0) {
            __hip_atomic_fetch_add(bar, 1u, __ATOMIC_RELEASE,
                                   __HIP_MEMORY_SCOPE_AGENT);
            const unsigned target = 64u * (unsigned)(t + 1);
            while (__hip_atomic_load(bar, __ATOMIC_ACQUIRE,
                                     __HIP_MEMORY_SCOPE_AGENT) < target)
                __builtin_amdgcn_s_sleep(2);
        }
        __syncthreads();
    }
}

extern "C" void kernel_launch(void* const* d_in, const int* in_sizes, int n_in,
                              void* d_out, int out_size, void* d_ws, size_t ws_size,
                              hipStream_t stream) {
    (void)in_sizes; (void)n_in; (void)out_size; (void)ws_size;
    const int T = 512, B = 64, I = 1024, H = 1024, L = 2;
    const int G4 = 4 * H;

    const float* x    = (const float*)d_in[0];  // [T,B,I]
    const float* W_ih = (const float*)d_in[1];  // [L,4H,I]
    const float* W_hh = (const float*)d_in[2];  // [L,4H,H]
    const float* bias = (const float*)d_in[3];  // [L,4H]
    const float* h0   = (const float*)d_in[4];  // [L,B,H]
    const float* c0   = (const float*)d_in[5];  // [L,B,H]

    float* out_seq = (float*)d_out;                       // [T,B,H]
    float* out_hn  = out_seq + (size_t)T * B * H;         // [L,B,H]
    float* out_cn  = out_hn  + (size_t)L * B * H;         // [L,B,H]

    char* ws = (char*)d_ws;
    _Float16* xb   = (_Float16*)ws; ws += (size_t)T * B * I * 2;   // x f16
    _Float16* s1b  = (_Float16*)ws; ws += (size_t)T * B * H * 2;   // layer0 out seq f16
    _Float16* wihb = (_Float16*)ws; ws += (size_t)G4 * I * 2;      // W_ih[layer] f16
    _Float16* whhb = (_Float16*)ws; ws += (size_t)G4 * H * 2;      // W_hh[layer] f16
    _Float16* gx   = (_Float16*)ws; ws += (size_t)T * B * G4 * 2;  // gate preacts f16
    _Float16* hb0  = (_Float16*)ws; ws += (size_t)B * H * 2;       // h ping
    _Float16* hb1  = (_Float16*)ws; ws += (size_t)B * H * 2;       // h pong
    unsigned* bar  = (unsigned*)ws;                                // step barrier

    const int CVT_T = 256;
    cvt_f32_to_f16<<<dim3(2048), dim3(CVT_T), 0, stream>>>(x, xb, T * B * I);

    for (int layer = 0; layer < L; ++layer) {
        cvt_f32_to_f16<<<dim3(1024), dim3(CVT_T), 0, stream>>>(
            W_ih + (size_t)layer * G4 * I, wihb, G4 * I);
        cvt_f32_to_f16<<<dim3(1024), dim3(CVT_T), 0, stream>>>(
            W_hh + (size_t)layer * G4 * H, whhb, G4 * H);
        cvt_f32_to_f16<<<dim3(64), dim3(CVT_T), 0, stream>>>(
            h0 + (size_t)layer * B * H, hb0, B * H);
        zero_u32<<<dim3(1), dim3(1), 0, stream>>>(bar);

        // gx = layer_input @ W_ih^T + b   (M = T*B, N = 4H, K = I)
        const _Float16* gemm_in = (layer == 0) ? xb : s1b;
        gemm_f16_wmma<<<dim3((T * B) / 128, G4 / 64), dim3(256), 0, stream>>>(
            gemm_in, wihb, bias + (size_t)layer * G4, gx, T * B, G4, I);

        lstm_scan_persistent<<<dim3(H / 16), dim3(1024), SMEM_BYTES, stream>>>(
            gx, whhb,
            c0 + (size_t)layer * B * H,
            hb0, hb1,
            out_cn + (size_t)layer * B * H,
            out_hn + (size_t)layer * B * H,
            (layer == L - 1) ? out_seq : nullptr,
            (layer == 0) ? s1b : nullptr,
            bar, T);
    }
}
